// Autoformer_41506563949110
// MI455X (gfx1250) — matrix operand order repaired and checked by reading, not compiled, not run on hardware
//
#include <hip/hip_runtime.h>
#include <math.h>

// ---------------------------------------------------------------- constants
#define NB    64
#define TI    192
#define TLAB  96
#define TOUT  96
#define VC    192     // V*C = C_IN
#define DM    512
#define DFF   2048
#define LSEQ  192     // both enc and dec sequence length
#define TOPK  5       // int(log(192)) = 5
#define KPAD  12      // (KMA-1)/2
#define MROWS (NB * LSEQ)   // 12288

typedef unsigned short u16;
typedef __attribute__((ext_vector_type(16))) __bf16 v16bf;
typedef __attribute__((ext_vector_type(8)))  float  v8f;

union Frag16 { v16bf v; u16 s[16]; };

__device__ __forceinline__ u16 f2bf(float f) {
  union { float f; unsigned u; } c; c.f = f;
  unsigned u = c.u;
  return (u16)((u + 0x7FFFu + ((u >> 16) & 1u)) >> 16);  // RNE
}

__device__ __forceinline__ float gelu_exact(float x) {
  return 0.5f * x * (1.0f + erff(x * 0.70710678118654752f));
}

// ---------------------------------------------------------------- weight cvt
__global__ void cvt_bf16_kernel(const float* __restrict__ in, u16* __restrict__ out, int n) {
  int i = blockIdx.x * blockDim.x + threadIdx.x;
  if (i < n) out[i] = f2bf(in[i]);
}

// ---------------------------------------------------------------- WMMA GEMM
// out[M,N] = act(A[M,K] @ W[K,N] + bias) + res ; bf16 inputs, f32 accumulate.
// Block: 256 threads = 8 waves; 128x64 output tile; each wave owns a 32x32
// region (2 A-frags x 2 B-frags -> 4 v_wmma per K-step, fragments reused 2x).
__global__ void __launch_bounds__(256)
gemm_bf16_wmma(const float* __restrict__ A, const u16* __restrict__ W,
               const float* __restrict__ bias, const float* __restrict__ res,
               float* __restrict__ out, int M, int K, int N, int do_gelu)
{
  __shared__ u16 As[128][32];  // [m][k]  8 KB
  __shared__ u16 Bs[64][32];   // transposed: [n][k]  4 KB
  const int bm   = blockIdx.x * 128;
  const int bn   = blockIdx.y * 64;
  const int tid  = threadIdx.x;
  const int lane = tid & 31;
  const int w    = tid >> 5;
  const int wm   = (w >> 1) * 32;   // wave row offset in tile (0..96)
  const int wn   = (w & 1) * 32;    // wave col offset in tile (0 or 32)
  const int half = lane >> 4;       // K-half selector
  const int l15  = lane & 15;

  v8f acc00 = {}, acc01 = {}, acc10 = {}, acc11 = {};

  for (int k0 = 0; k0 < K; k0 += 32) {
    const int more = (k0 + 32 < K);
    // stage A tile (coalesced along K) + prefetch next K-tile
    for (int i = tid; i < 128 * 32; i += 256) {
      int r = i >> 5, c = i & 31;
      const float* gp = &A[(size_t)(bm + r) * K + (k0 + c)];
      As[r][c] = f2bf(*gp);
      if (more) __builtin_prefetch(gp + 32, 0, 1);      // global_prefetch_b8
    }
    // stage B tile transposed (coalesced along N) + prefetch next K-tile
    for (int i = tid; i < 64 * 32; i += 256) {
      int nn = i & 63, kk = i >> 6;
      const u16* gp = &W[(size_t)(k0 + kk) * N + (bn + nn)];
      Bs[nn][kk] = *gp;
      if (more) __builtin_prefetch(gp + (size_t)32 * N, 0, 1);
    }
    __syncthreads();

    // A fragments: row = wm+l15 (+16); lanes 0-15: K 0-7 & 16-23 ; lanes 16-31: K 8-15 & 24-31
    Frag16 a0, a1, b0, b1;
    const u16* ar0 = &As[wm + l15][0];
    const u16* ar1 = &As[wm + 16 + l15][0];
#pragma unroll
    for (int j = 0; j < 8; ++j) {
      a0.s[j]     = ar0[half * 8 + j];
      a0.s[8 + j] = ar0[16 + half * 8 + j];
      a1.s[j]     = ar1[half * 8 + j];
      a1.s[8 + j] = ar1[16 + half * 8 + j];
    }
    // B fragments: col = l15 (+16); lanes 0-15: K 0-7 & 16-23 ; lanes 16-31: K 8-15 & 24-31
    const u16* bp0 = &Bs[wn + l15][0];
    const u16* bp1 = &Bs[wn + 16 + l15][0];
#pragma unroll
    for (int j = 0; j < 8; ++j) {
      b0.s[j] = bp0[half * 8 + j]; b0.s[8 + j] = bp0[16 + half * 8 + j];
      b1.s[j] = bp1[half * 8 + j]; b1.s[8 + j] = bp1[16 + half * 8 + j];
    }

    acc00 = __builtin_amdgcn_wmma_f32_16x16x32_bf16(false, a0.v, false, b0.v, (short)0, acc00, false, false);
    acc01 = __builtin_amdgcn_wmma_f32_16x16x32_bf16(false, a0.v, false, b1.v, (short)0, acc01, false, false);
    acc10 = __builtin_amdgcn_wmma_f32_16x16x32_bf16(false, a1.v, false, b0.v, (short)0, acc10, false, false);
    acc11 = __builtin_amdgcn_wmma_f32_16x16x32_bf16(false, a1.v, false, b1.v, (short)0, acc11, false, false);
    __syncthreads();
  }

  // C layout: col = l15, row = 8*half + r
#pragma unroll
  for (int r = 0; r < 8; ++r) {
    int row0 = bm + wm + half * 8 + r;
    int row1 = row0 + 16;
    int c0   = bn + wn + l15;
    int c1   = c0 + 16;
    float v00 = acc00[r], v01 = acc01[r], v10 = acc10[r], v11 = acc11[r];
    if (bias) {
      v00 += bias[c0]; v01 += bias[c1]; v10 += bias[c0]; v11 += bias[c1];
    }
    if (do_gelu) {
      v00 = gelu_exact(v00); v01 = gelu_exact(v01);
      v10 = gelu_exact(v10); v11 = gelu_exact(v11);
    }
    if (res) {
      v00 += res[(size_t)row0 * N + c0]; v01 += res[(size_t)row0 * N + c1];
      v10 += res[(size_t)row1 * N + c0]; v11 += res[(size_t)row1 * N + c1];
    }
    out[(size_t)row0 * N + c0] = v00;
    out[(size_t)row0 * N + c1] = v01;
    out[(size_t)row1 * N + c0] = v10;
    out[(size_t)row1 * N + c1] = v11;
  }
}

// ---------------------------------------------------------------- decomposition
// moving average window 25, edge-replicate padding. seas = x - m ; trend = m (opt accumulate)
__global__ void decomp_kernel(const float* __restrict__ X, float* __restrict__ seas,
                              float* __restrict__ trend, int total, int T, int C, int accum)
{
  int idx = blockIdx.x * blockDim.x + threadIdx.x;
  if (idx >= total) return;
  int c = idx % C;
  int t = (idx / C) % T;
  int n = idx / (C * T);
  const float* base = X + (size_t)n * T * C + c;
  float s = 0.f;
#pragma unroll
  for (int j = -KPAD; j <= KPAD; ++j) {
    int tt = t + j;
    tt = tt < 0 ? 0 : (tt >= T ? T - 1 : tt);
    s += base[(size_t)tt * C];
  }
  float m = s * (1.0f / 25.0f);
  if (seas)  seas[idx] = X[idx] - m;
  if (trend) { if (accum) trend[idx] += m; else trend[idx] = m; }
}

// mean over time axis: out[n,c] = mean_t X[n,t,c]
__global__ void time_mean_kernel(const float* __restrict__ X, float* __restrict__ out,
                                 int Nn, int T, int C)
{
  int idx = blockIdx.x * blockDim.x + threadIdx.x;
  if (idx >= Nn * C) return;
  int c = idx % C, n = idx / C;
  const float* p = X + (size_t)n * T * C + c;
  float s = 0.f;
  for (int t = 0; t < T; ++t) s += p[(size_t)t * C];
  out[idx] = s / (float)T;
}

__global__ void sub_colmean_kernel(float* __restrict__ X, const float* __restrict__ cm,
                                   int total, int T, int C)
{
  int idx = blockIdx.x * blockDim.x + threadIdx.x;
  if (idx >= total) return;
  int c = idx % C;
  int n = idx / (C * T);
  X[idx] -= cm[n * C + c];
}

// ---------------------------------------------------------------- decoder inits
__global__ void build_inits_kernel(const float* __restrict__ seas, const float* __restrict__ trend0,
                                   const float* __restrict__ meanX,
                                   float* __restrict__ seasInit, float* __restrict__ trendInit)
{
  int idx = blockIdx.x * blockDim.x + threadIdx.x;
  if (idx >= NB * LSEQ * VC) return;
  int c = idx % VC;
  int t = (idx / VC) % LSEQ;
  int n = idx / (VC * LSEQ);
  if (t < TLAB) {
    int src = ((n * TI) + (TI - TLAB + t)) * VC + c;
    seasInit[idx]  = seas[src];
    trendInit[idx] = trend0[src];
  } else {
    seasInit[idx]  = 0.f;
    trendInit[idx] = meanX[n * VC + c];
  }
}

__global__ void build_mark_kernel(const int* __restrict__ xF, const int* __restrict__ yF,
                                  int* __restrict__ mark)
{
  int idx = blockIdx.x * blockDim.x + threadIdx.x;
  if (idx >= NB * LSEQ * 4) return;
  int f = idx & 3;
  int t = (idx >> 2) % LSEQ;
  int n = idx / (LSEQ * 4);
  mark[idx] = (t < TLAB) ? xF[((n * TI) + (TI - TLAB + t)) * 4 + f]
                         : yF[((n * TOUT) + (t - TLAB)) * 4 + f];
}

// ---------------------------------------------------------------- circular conv (+temporal +base)
// out[n,t,o] = sum_k sum_i X[n,(t+k-1)%T,i] * Wc[o,i,k]  (+ temporal(mark) if mark) (+ base)
__global__ void conv_circ_kernel(const float* __restrict__ X, const float* __restrict__ Wc,
                                 const int* __restrict__ mark, const float* __restrict__ base,
                                 float* __restrict__ out, int Nn, int T, int Cin, int Cout)
{
  int idx = blockIdx.x * blockDim.x + threadIdx.x;
  if (idx >= Nn * T * Cout) return;
  int o = idx % Cout;
  int t = (idx / Cout) % T;
  int n = idx / (Cout * T);
  int tm = (t == 0) ? T - 1 : t - 1;
  int tp = (t == T - 1) ? 0 : t + 1;
  const float* x0 = X + ((size_t)n * T + tm) * Cin;
  const float* x1 = X + ((size_t)n * T + t ) * Cin;
  const float* x2 = X + ((size_t)n * T + tp) * Cin;
  const float* wp = Wc + (size_t)o * Cin * 3;
  float acc = 0.f;
  for (int i = 0; i < Cin; ++i)
    acc = fmaf(x0[i], wp[3 * i + 0], fmaf(x1[i], wp[3 * i + 1], fmaf(x2[i], wp[3 * i + 2], acc)));
  if (mark) {  // fixed sinusoidal temporal embedding, d=512 tables, 4 features summed
    const int* m = mark + ((size_t)n * T + t) * 4;
    float div = expf(-(float)(o & ~1) * (9.2103403719761836f / 512.0f));
    float s;
    if (o & 1) s = cosf(m[3] * div) + cosf(m[2] * div) + cosf(m[1] * div) + cosf(m[0] * div);
    else       s = sinf(m[3] * div) + sinf(m[2] * div) + sinf(m[1] * div) + sinf(m[0] * div);
    acc += s;
  }
  if (base) acc += base[idx];
  out[idx] = acc;
}

// ---------------------------------------------------------------- autocorrelation
// Mv[b,tau] = (1/D) * sum_t dot(q[b,(t+tau)%L,:], k[b,t,:])   (== mean over (h,e) of irfft(qf*conj(kf)))
__global__ void __launch_bounds__(256)
autocorr_mean_kernel(const float* __restrict__ Q, const float* __restrict__ Kk,
                     float* __restrict__ Mv, int L, int D)
{
  int b   = blockIdx.x;
  int tau = blockIdx.y;
  const float* qb = Q  + (size_t)b * L * D;
  const float* kb = Kk + (size_t)b * L * D;
  float acc = 0.f;
  for (int t = 0; t < L; ++t) {
    int tq = t + tau; if (tq >= L) tq -= L;
    const float* qr = qb + (size_t)tq * D;
    const float* kr = kb + (size_t)t  * D;
    for (int d = threadIdx.x; d < D; d += 256) acc += qr[d] * kr[d];
  }
  __shared__ float red[256];
  red[threadIdx.x] = acc;
  __syncthreads();
  for (int s = 128; s > 0; s >>= 1) {
    if (threadIdx.x < s) red[threadIdx.x] += red[threadIdx.x + s];
    __syncthreads();
  }
  if (threadIdx.x == 0) Mv[b * L + tau] = red[0] / (float)D;
}

__global__ void topk_softmax_kernel(const float* __restrict__ Mv, float* __restrict__ wts,
                                    int* __restrict__ dly, int L)
{
  int b = blockIdx.x;
  __shared__ float vals[256];
  for (int i = threadIdx.x; i < L; i += blockDim.x) vals[i] = Mv[b * L + i];
  __syncthreads();
  if (threadIdx.x == 0) {
    float wv[TOPK]; int dv[TOPK];
    for (int k = 0; k < TOPK; ++k) {
      float best = -3.4e38f; int bi = 0;
      for (int i = 0; i < L; ++i) if (vals[i] > best) { best = vals[i]; bi = i; }
      wv[k] = best; dv[k] = bi; vals[bi] = -3.4e38f;
    }
    float mx = wv[0];
    float ex[TOPK], se = 0.f;
    for (int k = 0; k < TOPK; ++k) { ex[k] = expf(wv[k] - mx); se += ex[k]; }
    for (int k = 0; k < TOPK; ++k) {
      wts[b * TOPK + k] = ex[k] / se;
      dly[b * TOPK + k] = dv[k];
    }
  }
}

// out[b,l,d] = sum_k w[b,k] * V[b,(l+delay[b,k])%L, d]
__global__ void agg_kernel(const float* __restrict__ V, const float* __restrict__ wts,
                           const int* __restrict__ dly, float* __restrict__ out,
                           int L, int D, int total)
{
  int idx = blockIdx.x * blockDim.x + threadIdx.x;
  if (idx >= total) return;
  int d = idx % D;
  int l = (idx / D) % L;
  int b = idx / (D * L);
  float acc = 0.f;
#pragma unroll
  for (int k = 0; k < TOPK; ++k) {
    int t = l + dly[b * TOPK + k]; if (t >= L) t -= L;
    acc += wts[b * TOPK + k] * V[((size_t)b * L + t) * D + d];
  }
  out[idx] = acc;
}

// ---------------------------------------------------------------- layernorm (last dim)
__global__ void __launch_bounds__(256)
ln_kernel(const float* __restrict__ X, const float* __restrict__ g, const float* __restrict__ bta,
          float* __restrict__ out, int D)
{
  int row = blockIdx.x;
  const float* x = X + (size_t)row * D;
  __shared__ float red[256];
  float s = 0.f;
  for (int d = threadIdx.x; d < D; d += 256) s += x[d];
  red[threadIdx.x] = s; __syncthreads();
  for (int st = 128; st > 0; st >>= 1) { if (threadIdx.x < st) red[threadIdx.x] += red[threadIdx.x + st]; __syncthreads(); }
  float mu = red[0] / (float)D;
  __syncthreads();
  float v = 0.f;
  for (int d = threadIdx.x; d < D; d += 256) { float dv = x[d] - mu; v += dv * dv; }
  red[threadIdx.x] = v; __syncthreads();
  for (int st = 128; st > 0; st >>= 1) { if (threadIdx.x < st) red[threadIdx.x] += red[threadIdx.x + st]; __syncthreads(); }
  float inv = rsqrtf(red[0] / (float)D + 1e-5f);
  for (int d = threadIdx.x; d < D; d += 256)
    out[(size_t)row * D + d] = (x[d] - mu) * inv * g[d] + bta[d];
}

// ---------------------------------------------------------------- final slice
__global__ void final_out_kernel(const float* __restrict__ trend, const float* __restrict__ seasn,
                                 float* __restrict__ out)
{
  int idx = blockIdx.x * blockDim.x + threadIdx.x;
  if (idx >= NB * TOUT * VC) return;
  int c = idx % VC;
  int t = (idx / VC) % TOUT;
  int n = idx / (VC * TOUT);
  int src = ((n * LSEQ) + (LSEQ - TOUT + t)) * VC + c;
  out[idx] = trend[src] + seasn[src];
}

// ================================================================ host driver
extern "C" void kernel_launch(void* const* d_in, const int* in_sizes, int n_in,
                              void* d_out, int out_size, void* d_ws, size_t ws_size,
                              hipStream_t stream)
{
  (void)in_sizes; (void)n_in; (void)out_size; (void)ws_size;

  const float* x     = (const float*)d_in[0];   // (64,192,64,3) == xe (64,192,192)
  const int*   xF    = (const int*)  d_in[2];
  const int*   yF    = (const int*)  d_in[3];
  const float* embE  = (const float*)d_in[4];
  const float* embD  = (const float*)d_in[5];
  const float* encW[4]  = {(const float*)d_in[6],(const float*)d_in[7],(const float*)d_in[8],(const float*)d_in[9]};
  const float* encBv[4] = {(const float*)d_in[10],(const float*)d_in[11],(const float*)d_in[12],(const float*)d_in[13]};
  const float* encW1 = (const float*)d_in[14];
  const float* encW2 = (const float*)d_in[15];
  const float* encLnG = (const float*)d_in[16];
  const float* encLnB = (const float*)d_in[17];
  const float* decSW[4] = {(const float*)d_in[18],(const float*)d_in[19],(const float*)d_in[20],(const float*)d_in[21]};
  const float* decSB[4] = {(const float*)d_in[22],(const float*)d_in[23],(const float*)d_in[24],(const float*)d_in[25]};
  const float* decCW[4] = {(const float*)d_in[26],(const float*)d_in[27],(const float*)d_in[28],(const float*)d_in[29]};
  const float* decCB[4] = {(const float*)d_in[30],(const float*)d_in[31],(const float*)d_in[32],(const float*)d_in[33]};
  const float* decW1  = (const float*)d_in[34];
  const float* decW2  = (const float*)d_in[35];
  const float* trendW = (const float*)d_in[36];
  const float* decLnG = (const float*)d_in[37];
  const float* decLnB = (const float*)d_in[38];
  const float* projW  = (const float*)d_in[39];
  const float* projB  = (const float*)d_in[40];

  // ---- workspace carve
  char* wp = (char*)d_ws;
  auto take = [&](size_t bytes) -> char* {
    char* r = wp; wp += (bytes + 255) & ~(size_t)255; return r;
  };
  const int DD = DM * DM;           // 262144
  const int DH = DM * DFF;          // 1048576
  u16* wbEnc[4]; for (int i = 0; i < 4; ++i) wbEnc[i] = (u16*)take((size_t)2 * DD * sizeof(u16));
  u16* wbEncW1 = (u16*)take((size_t)2 * DH * sizeof(u16));
  u16* wbEncW2 = (u16*)take((size_t)2 * DH * sizeof(u16));
  u16* wbDecS[4]; for (int i = 0; i < 4; ++i) wbDecS[i] = (u16*)take((size_t)DD * sizeof(u16));
  u16* wbDecC[4]; for (int i = 0; i < 4; ++i) wbDecC[i] = (u16*)take((size_t)DD * sizeof(u16));
  u16* wbDecW1 = (u16*)take((size_t)DH * sizeof(u16));
  u16* wbDecW2 = (u16*)take((size_t)DH * sizeof(u16));
  u16* wbProj  = (u16*)take((size_t)DM * VC * sizeof(u16));

  const int SZ_SMALL = NB * LSEQ * VC;   // 2,359,296
  const int SZ_ACT   = NB * LSEQ * DM;   // 6,291,456
  const int SZ_H     = NB * LSEQ * DFF;  // 25,165,824
  float* seasB    = (float*)take((size_t)SZ_SMALL * 4);
  float* trend0B  = (float*)take((size_t)SZ_SMALL * 4);
  float* seasInit = (float*)take((size_t)SZ_SMALL * 4);
  float* trendInit= (float*)take((size_t)SZ_SMALL * 4);
  float* trendB   = (float*)take((size_t)SZ_SMALL * 4);
  float* meanX    = (float*)take((size_t)NB * VC * 4);
  float* colmean  = (float*)take((size_t)NB * DM * 4);
  int*   markDec  = (int*)  take((size_t)NB * LSEQ * 4 * 4);
  float* Mv       = (float*)take((size_t)NB * LSEQ * 4);
  float* wts      = (float*)take((size_t)NB * TOPK * 4);
  int*   dly      = (int*)  take((size_t)NB * TOPK * 4);
  float* encA  = (float*)take((size_t)SZ_ACT * 4);
  float* encB_ = (float*)take((size_t)SZ_ACT * 4);
  float* decA  = (float*)take((size_t)SZ_ACT * 4);
  float* decB_ = (float*)take((size_t)SZ_ACT * 4);
  float* Qb    = (float*)take((size_t)SZ_ACT * 4);
  float* Kb    = (float*)take((size_t)SZ_ACT * 4);
  float* Vb    = (float*)take((size_t)SZ_ACT * 4);
  float* Tt    = (float*)take((size_t)SZ_ACT * 4);
  float* TS    = (float*)take((size_t)SZ_ACT * 4);   // t1+t2+t3 accumulator
  float* Hb    = (float*)take((size_t)SZ_H * 4);

  auto cvt = [&](const float* src, u16* dst, int n) {
    cvt_bf16_kernel<<<(n + 255) / 256, 256, 0, stream>>>(src, dst, n);
  };
  auto gemm = [&](const float* A, const u16* W, const float* bias, const float* res,
                  float* out, int M, int K, int N, int act) {
    dim3 grid(M / 128, N / 64);
    gemm_bf16_wmma<<<grid, 256, 0, stream>>>(A, W, bias, res, out, M, K, N, act);
  };
  auto autocorr = [&](const float* Qp, const float* Kp, const float* Vp, float* outp) {
    autocorr_mean_kernel<<<dim3(NB, LSEQ), 256, 0, stream>>>(Qp, Kp, Mv, LSEQ, DM);
    topk_softmax_kernel<<<NB, 256, 0, stream>>>(Mv, wts, dly, LSEQ);
    int tot = NB * LSEQ * DM;
    agg_kernel<<<(tot + 255) / 256, 256, 0, stream>>>(Vp, wts, dly, outp, LSEQ, DM, tot);
  };
  auto my_ln = [&](float*& cur, float*& alt, const float* g, const float* b) {
    ln_kernel<<<MROWS, 256, 0, stream>>>(cur, g, b, alt, DM);
    time_mean_kernel<<<(NB * DM + 255) / 256, 256, 0, stream>>>(alt, colmean, NB, LSEQ, DM);
    sub_colmean_kernel<<<(NB * LSEQ * DM + 255) / 256, 256, 0, stream>>>(alt, colmean, NB * LSEQ * DM, LSEQ, DM);
    float* t = cur; cur = alt; alt = t;
  };

  // ---- 0. convert weights to bf16
  for (int i = 0; i < 4; ++i) cvt(encW[i], wbEnc[i], 2 * DD);
  cvt(encW1, wbEncW1, 2 * DH);
  cvt(encW2, wbEncW2, 2 * DH);
  for (int i = 0; i < 4; ++i) cvt(decSW[i], wbDecS[i], DD);
  for (int i = 0; i < 4; ++i) cvt(decCW[i], wbDecC[i], DD);
  cvt(decW1, wbDecW1, DH);
  cvt(decW2, wbDecW2, DH);
  cvt(projW, wbProj, DM * VC);

  // ---- 1. series decomposition of input + decoder inits
  time_mean_kernel<<<(NB * VC + 255) / 256, 256, 0, stream>>>(x, meanX, NB, TI, VC);
  decomp_kernel<<<(SZ_SMALL + 255) / 256, 256, 0, stream>>>(x, seasB, trend0B, SZ_SMALL, TI, VC, 0);
  build_inits_kernel<<<(SZ_SMALL + 255) / 256, 256, 0, stream>>>(seasB, trend0B, meanX, seasInit, trendInit);
  build_mark_kernel<<<(NB * LSEQ * 4 + 255) / 256, 256, 0, stream>>>(xF, yF, markDec);

  // ---- 2. encoder embedding: circ_conv + temporal
  conv_circ_kernel<<<(SZ_ACT + 255) / 256, 256, 0, stream>>>(x, embE, xF, nullptr, encA, NB, TI, VC, DM);
  float* enc = encA; float* encAlt = encB_;

  // ---- 3. encoder layers
  for (int i = 0; i < 2; ++i) {
    const u16* Wq = wbEnc[0] + (size_t)i * DD;
    const u16* Wk = wbEnc[1] + (size_t)i * DD;
    const u16* Wv = wbEnc[2] + (size_t)i * DD;
    const u16* Wo = wbEnc[3] + (size_t)i * DD;
    const float* bq = encBv[0] + i * DM; const float* bk = encBv[1] + i * DM;
    const float* bv = encBv[2] + i * DM; const float* bo = encBv[3] + i * DM;
    gemm(enc, Wq, bq, nullptr, Qb, MROWS, DM, DM, 0);
    gemm(enc, Wk, bk, nullptr, Kb, MROWS, DM, DM, 0);
    gemm(enc, Wv, bv, nullptr, Vb, MROWS, DM, DM, 0);
    autocorr(Qb, Kb, Vb, Tt);
    gemm(Tt, Wo, bo, /*res=*/enc, enc, MROWS, DM, DM, 0);          // enc += attn(enc)
    decomp_kernel<<<(SZ_ACT + 255) / 256, 256, 0, stream>>>(enc, encAlt, nullptr, SZ_ACT, LSEQ, DM, 0);
    { float* t = enc; enc = encAlt; encAlt = t; }
    gemm(enc, wbEncW1 + (size_t)i * DH, nullptr, nullptr, Hb, MROWS, DM, DFF, 1);   // gelu(enc@W1)
    gemm(Hb,  wbEncW2 + (size_t)i * DH, nullptr, /*res=*/enc, Tt, MROWS, DFF, DM, 0);
    decomp_kernel<<<(SZ_ACT + 255) / 256, 256, 0, stream>>>(Tt, encAlt, nullptr, SZ_ACT, LSEQ, DM, 0);
    { float* t = enc; enc = encAlt; encAlt = t; }
  }
  my_ln(enc, encAlt, encLnG, encLnB);

  // ---- 4. decoder embedding
  conv_circ_kernel<<<(SZ_ACT + 255) / 256, 256, 0, stream>>>(seasInit, embD, markDec, nullptr, decA, NB, LSEQ, VC, DM);
  float* dec = decA; float* decAlt = decB_;

  // self attention
  gemm(dec, wbDecS[0], decSB[0], nullptr, Qb, MROWS, DM, DM, 0);
  gemm(dec, wbDecS[1], decSB[1], nullptr, Kb, MROWS, DM, DM, 0);
  gemm(dec, wbDecS[2], decSB[2], nullptr, Vb, MROWS, DM, DM, 0);
  autocorr(Qb, Kb, Vb, Tt);
  gemm(Tt, wbDecS[3], decSB[3], /*res=*/dec, dec, MROWS, DM, DM, 0);
  decomp_kernel<<<(SZ_ACT + 255) / 256, 256, 0, stream>>>(dec, decAlt, TS, SZ_ACT, LSEQ, DM, 0);  // t1
  { float* t = dec; dec = decAlt; decAlt = t; }

  // cross attention (k, v from encoder)
  gemm(dec, wbDecC[0], decCB[0], nullptr, Qb, MROWS, DM, DM, 0);
  gemm(enc, wbDecC[1], decCB[1], nullptr, Kb, MROWS, DM, DM, 0);
  gemm(enc, wbDecC[2], decCB[2], nullptr, Vb, MROWS, DM, DM, 0);
  autocorr(Qb, Kb, Vb, Tt);
  gemm(Tt, wbDecC[3], decCB[3], /*res=*/dec, dec, MROWS, DM, DM, 0);
  decomp_kernel<<<(SZ_ACT + 255) / 256, 256, 0, stream>>>(dec, decAlt, TS, SZ_ACT, LSEQ, DM, 1);  // += t2
  { float* t = dec; dec = decAlt; decAlt = t; }

  // FFN
  gemm(dec, wbDecW1, nullptr, nullptr, Hb, MROWS, DM, DFF, 1);
  gemm(Hb,  wbDecW2, nullptr, /*res=*/dec, Tt, MROWS, DFF, DM, 0);
  decomp_kernel<<<(SZ_ACT + 255) / 256, 256, 0, stream>>>(Tt, decAlt, TS, SZ_ACT, LSEQ, DM, 1);   // += t3
  { float* t = dec; dec = decAlt; decAlt = t; }

  // ---- 5. trend path: trend = trend_init + circ_conv(t1+t2+t3, dec_trend_w)
  conv_circ_kernel<<<(SZ_SMALL + 255) / 256, 256, 0, stream>>>(TS, trendW, nullptr, trendInit, trendB, NB, LSEQ, DM, VC);

  // ---- 6. seasonal path: my_ln then projection
  my_ln(dec, decAlt, decLnG, decLnB);
  gemm(dec, wbProj, projB, nullptr, Tt, MROWS, DM, VC, 0);   // (N,192,192) seasonal

  // ---- 7. output slice
  final_out_kernel<<<(NB * TOUT * VC + 255) / 256, 256, 0, stream>>>(trendB, Tt, (float*)d_out);
}
